// TransformerBlock_46883863003628
// MI455X (gfx1250) — compile-verified, matrix operand order checked by reading, not run on hardware
//
#include <hip/hip_runtime.h>
#include <hip/hip_bf16.h>

// ---------------- problem constants (fixed by the reference) ----------------
#define Bsz 4
#define Tt  2048
#define Cc  768
#define Hh  12
#define Dd  64
#define Mrows (Bsz * Tt)   // 8192
#define EPSf 1e-5f
#define NEG_BIG -3.0e38f

// ---------------- WMMA types & helpers ----------------
typedef __attribute__((ext_vector_type(16))) __bf16 v16bf;
typedef __attribute__((ext_vector_type(8)))  float  v8f;

union Frag16 { v16bf v; unsigned u[8]; };

__device__ __forceinline__ v8f wmma_bf16(v16bf a, v16bf b, v8f c) {
  // D = A(16x32 bf16) * B(32x16 bf16) + C(16x16 f32)
  return __builtin_amdgcn_wmma_f32_16x16x32_bf16(
      /*neg_a=*/false, a, /*neg_b=*/false, b,
      /*c_mod=*/(short)0, c, /*reuse_a=*/false, /*reuse_b=*/false);
}

// Loads a 16x32 bf16 fragment in the CDNA5 A-matrix layout from a row-major
// matrix (rows = M for A-operand, rows = N for B-operand since B == (B^T as A)).
// Requires k0 % 32 == 0 and even ld (dword-aligned rows).
__device__ __forceinline__ v16bf load_frag(const __bf16* base, int ld, int row0, int k0) {
  const int lane = threadIdx.x & 31;
  const int half = lane >> 4;
  const int idx  = lane & 15;
  const unsigned* rp = reinterpret_cast<const unsigned*>(base + (size_t)(row0 + idx) * (size_t)ld);
  const int kb = (k0 >> 1) + half * 4;   // dword index of first K pair
  Frag16 f;
#pragma unroll
  for (int v = 0; v < 4; ++v) {
    f.u[v]     = rp[kb + v];       // K = k0 + half*8 + 2v, +1
    f.u[4 + v] = rp[kb + 8 + v];   // K = +16
  }
  return f.v;
}

__device__ __forceinline__ float half_max16(float v) {
#pragma unroll
  for (int off = 1; off < 16; off <<= 1) v = fmaxf(v, __shfl_xor(v, off));
  return v;
}
__device__ __forceinline__ float half_sum16(float v) {
#pragma unroll
  for (int off = 1; off < 16; off <<= 1) v += __shfl_xor(v, off);
  return v;
}

// ---------------- weight convert: W (K x N, f32) -> Wt (N x K, bf16) ----------------
__global__ __launch_bounds__(256) void wcvt_k(const float* __restrict__ W,
                                              __bf16* __restrict__ Wt,
                                              int K, int N) {
  int i = blockIdx.x * 256 + threadIdx.x;
  if (i >= K * N) return;
  int n = i / K, k = i - n * K;
  Wt[(size_t)n * K + k] = (__bf16)W[(size_t)k * N + n];
}

// ---------------- LayerNorm: fp32 in -> bf16 normalized out ----------------
__global__ __launch_bounds__(256) void ln_k(const float* __restrict__ x,
                                            const float* __restrict__ g,
                                            const float* __restrict__ bt,
                                            __bf16* __restrict__ out) {
  __shared__ float red[256];
  const int row = blockIdx.x;
  const int tid = threadIdx.x;
  const float* xr = x + (size_t)row * Cc;
  float v0 = xr[tid], v1 = xr[tid + 256], v2 = xr[tid + 512];

  red[tid] = v0 + v1 + v2;
  __syncthreads();
#pragma unroll
  for (int off = 128; off > 0; off >>= 1) {
    if (tid < off) red[tid] += red[tid + off];
    __syncthreads();
  }
  const float mu = red[0] * (1.0f / Cc);
  __syncthreads();

  float d0 = v0 - mu, d1 = v1 - mu, d2 = v2 - mu;
  red[tid] = d0 * d0 + d1 * d1 + d2 * d2;
  __syncthreads();
#pragma unroll
  for (int off = 128; off > 0; off >>= 1) {
    if (tid < off) red[tid] += red[tid + off];
    __syncthreads();
  }
  const float rstd = rsqrtf(red[0] * (1.0f / Cc) + EPSf);

  __bf16* orow = out + (size_t)row * Cc;
  orow[tid]       = (__bf16)(d0 * rstd * g[tid]       + bt[tid]);
  orow[tid + 256] = (__bf16)(d1 * rstd * g[tid + 256] + bt[tid + 256]);
  orow[tid + 512] = (__bf16)(d2 * rstd * g[tid + 512] + bt[tid + 512]);
}

// ---------------- generic WMMA GEMM: (Mrows x K) bf16 @ Wt(N x K) bf16 ----------------
// Workgroup = 4 waves = 128x64 output tile; each wave owns 32x64
// (2 A-frags + 4 B-frags -> 8 WMMAs per K-step; 8 independent accum chains).
// MODE 0: out bf16 scattered to (B,H,T,D)   (Q, K tensors)
// MODE 1: out bf16 scattered to (B,H,D,T)   (V transposed for the PV matmul)
// MODE 2: outf = resid + acc + bias (fp32, row-major MrowsxN; requires N==Cc)
// MODE 3: outb = gelu(acc + bias) bf16 row-major MrowsxN
template <int N, int K, int MODE>
__global__ __launch_bounds__(128) void gemm_k(const __bf16* __restrict__ A,
                                              const __bf16* __restrict__ Wt,
                                              const float* __restrict__ bias,
                                              const float* __restrict__ resid,
                                              float* __restrict__ outf,
                                              __bf16* __restrict__ outb) {
  const int wave = threadIdx.x >> 5;
  const int lane = threadIdx.x & 31;
  const int half = lane >> 4;
  const int idx  = lane & 15;
  const int m0 = blockIdx.x * 128 + wave * 32;
  const int n0 = blockIdx.y * 64;

  v8f acc[2][4] = {};
  for (int k0 = 0; k0 < K; k0 += 32) {
    v16bf af0 = load_frag(A, K, m0,      k0);
    v16bf af1 = load_frag(A, K, m0 + 16, k0);
#pragma unroll
    for (int nt = 0; nt < 4; ++nt) {
      v16bf bf = load_frag(Wt, K, n0 + nt * 16, k0);
      acc[0][nt] = wmma_bf16(af0, bf, acc[0][nt]);
      acc[1][nt] = wmma_bf16(af1, bf, acc[1][nt]);
    }
  }

#pragma unroll
  for (int mi = 0; mi < 2; ++mi) {
#pragma unroll
    for (int nt = 0; nt < 4; ++nt) {
      const int n = n0 + nt * 16 + idx;
      const float bv = bias[n];
#pragma unroll
      for (int r = 0; r < 8; ++r) {
        const int m = m0 + mi * 16 + r + 8 * half;
        const float val = acc[mi][nt][r] + bv;
        if (MODE == 0 || MODE == 1) {
          const int bb = m >> 11, t = m & (Tt - 1);
          const int hh = n >> 6,  d = n & (Dd - 1);
          const size_t off = (MODE == 0)
              ? ((((size_t)bb * Hh + hh) * Tt + t) * Dd + d)
              : ((((size_t)bb * Hh + hh) * Dd + d) * Tt + t);
          outb[off] = (__bf16)val;
        } else if (MODE == 2) {
          outf[(size_t)m * Cc + n] = resid[(size_t)m * Cc + n] + val;
        } else { // MODE 3: exact GELU
          const float gv = 0.5f * val * (1.0f + erff(val * 0.70710678118654752f));
          outb[(size_t)m * N + n] = (__bf16)gv;
        }
      }
    }
  }
}

// ---------------- causal flash attention ----------------
// q,k: (B,H,T,D) bf16 ; vt: (B,H,D,T) bf16 ; o: (B,T,C) bf16
// grid: (T/64, B*H), block: 128 (4 waves, 16 query rows each)
__global__ __launch_bounds__(128) void attn_k(const __bf16* __restrict__ q,
                                              const __bf16* __restrict__ kk,
                                              const __bf16* __restrict__ vt,
                                              __bf16* __restrict__ o) {
  const int qb = blockIdx.x;   // 0..31
  const int bh = blockIdx.y;   // 0..47
  const int b  = bh / Hh;
  const int hh = bh - b * Hh;
  const int wave = threadIdx.x >> 5;
  const int lane = threadIdx.x & 31;
  const int half = lane >> 4;
  const int idx  = lane & 15;

  const __bf16* qp = q  + (size_t)bh * Tt * Dd;
  const __bf16* kp = kk + (size_t)bh * Tt * Dd;
  const __bf16* vp = vt + (size_t)bh * Dd * Tt;

  const int q0 = qb * 64 + wave * 16;
  const v16bf qf0 = load_frag(qp, Dd, q0, 0);
  const v16bf qf1 = load_frag(qp, Dd, q0, 32);

  v8f acc[4] = {};
  float rmax[8], rsum[8];
#pragma unroll
  for (int r = 0; r < 8; ++r) { rmax[r] = NEG_BIG; rsum[r] = 0.0f; }

  __shared__ __bf16 pl[4][16 * 32];  // per-wave P staging (C-layout -> A-layout)

  const int nkb = 2 * qb + 2;        // uniform across waves -> barriers are safe
  for (int kt = 0; kt < nkb; ++kt) {
    const int key0 = kt * 32;

    // scores S = (Q K^T) / sqrt(D), causal-masked, held as two 16x16 f32 tiles
    float sv[2][8];
#pragma unroll
    for (int s = 0; s < 2; ++s) {
      v8f sc = {};
      v16bf kf0 = load_frag(kp, Dd, key0 + s * 16, 0);
      v16bf kf1 = load_frag(kp, Dd, key0 + s * 16, 32);
      sc = wmma_bf16(qf0, kf0, sc);
      sc = wmma_bf16(qf1, kf1, sc);
      const int kg = key0 + s * 16 + idx;
#pragma unroll
      for (int r = 0; r < 8; ++r) {
        const int qg = q0 + r + 8 * half;
        sv[s][r] = (kg <= qg) ? (sc[r] * 0.125f) : NEG_BIG;
      }
    }

    // online softmax (per row = per (r, half))
    float alpha[8];
#pragma unroll
    for (int r = 0; r < 8; ++r) {
      float mloc = half_max16(fmaxf(sv[0][r], sv[1][r]));
      const float nm = fmaxf(rmax[r], mloc);
      alpha[r] = __expf(rmax[r] - nm);
      rmax[r] = nm;
      const float p0 = __expf(sv[0][r] - nm);
      const float p1 = __expf(sv[1][r] - nm);
      rsum[r] = rsum[r] * alpha[r] + half_sum16(p0 + p1);
      const int mrow = r + 8 * half;
      pl[wave][mrow * 32 + idx]      = (__bf16)p0;
      pl[wave][mrow * 32 + 16 + idx] = (__bf16)p1;
    }
#pragma unroll
    for (int nt = 0; nt < 4; ++nt)
#pragma unroll
      for (int r = 0; r < 8; ++r) acc[nt][r] *= alpha[r];

    __syncthreads();  // P tile visible across lanes of the wave
    const v16bf pf = load_frag(&pl[wave][0], 32, 0, 0);
#pragma unroll
    for (int nt = 0; nt < 4; ++nt) {
      v16bf vf = load_frag(vp, Tt, nt * 16, key0);  // rows = d, K = key (contiguous)
      acc[nt] = wmma_bf16(pf, vf, acc[nt]);
    }
    __syncthreads();  // protect pl before next iteration's writes
  }

  // epilogue: O / rowsum -> bf16 at (b, t, hh*64 + d)
#pragma unroll
  for (int nt = 0; nt < 4; ++nt) {
#pragma unroll
    for (int r = 0; r < 8; ++r) {
      const int t = q0 + r + 8 * half;
      const int c = hh * Dd + nt * 16 + idx;
      o[((size_t)b * Tt + t) * Cc + c] = (__bf16)(acc[nt][r] / rsum[r]);
    }
  }
}

// ---------------- host-side orchestration ----------------
extern "C" void kernel_launch(void* const* d_in, const int* in_sizes, int n_in,
                              void* d_out, int out_size, void* d_ws, size_t ws_size,
                              hipStream_t stream) {
  const float* x   = (const float*)d_in[0];
  const float* Wq  = (const float*)d_in[1];
  const float* bq  = (const float*)d_in[2];
  const float* Wk  = (const float*)d_in[3];
  const float* bk  = (const float*)d_in[4];
  const float* Wv  = (const float*)d_in[5];
  const float* bv  = (const float*)d_in[6];
  const float* Wo  = (const float*)d_in[7];
  const float* bo  = (const float*)d_in[8];
  const float* g1  = (const float*)d_in[9];
  const float* b1  = (const float*)d_in[10];
  const float* g2  = (const float*)d_in[11];
  const float* b2  = (const float*)d_in[12];
  const float* W1  = (const float*)d_in[13];
  const float* bm1 = (const float*)d_in[14];
  const float* W2  = (const float*)d_in[15];
  const float* bm2 = (const float*)d_in[16];
  float* out = (float*)d_out;

  // workspace carve-up (256B aligned)
  char* ws = (char*)d_ws;
  size_t off = 0;
  auto alloc = [&](size_t bytes) -> void* {
    void* p = ws + off;
    off = (off + bytes + 255) & ~(size_t)255;
    return p;
  };
  const size_t CC  = (size_t)Cc * Cc;
  const size_t C4C = (size_t)Cc * 4 * Cc;
  const size_t MC  = (size_t)Mrows * Cc;
  const size_t M4C = (size_t)Mrows * 4 * Cc;

  __bf16* Wqt = (__bf16*)alloc(CC * 2);
  __bf16* Wkt = (__bf16*)alloc(CC * 2);
  __bf16* Wvt = (__bf16*)alloc(CC * 2);
  __bf16* Wot = (__bf16*)alloc(CC * 2);
  __bf16* W1t = (__bf16*)alloc(C4C * 2);
  __bf16* W2t = (__bf16*)alloc(C4C * 2);
  __bf16* h1b = (__bf16*)alloc(MC * 2);
  __bf16* qbf = (__bf16*)alloc(MC * 2);
  __bf16* kbf = (__bf16*)alloc(MC * 2);
  __bf16* vtb = (__bf16*)alloc(MC * 2);
  __bf16* atb = (__bf16*)alloc(MC * 2);
  float*  x2  = (float*) alloc(MC * 4);
  __bf16* h2b = (__bf16*)alloc(MC * 2);
  __bf16* mb  = (__bf16*)alloc(M4C * 2);
  (void)ws_size; (void)in_sizes; (void)n_in; (void)out_size;

  // 1) weights -> transposed bf16
  wcvt_k<<<(int)((CC  + 255) / 256), 256, 0, stream>>>(Wq, Wqt, Cc, Cc);
  wcvt_k<<<(int)((CC  + 255) / 256), 256, 0, stream>>>(Wk, Wkt, Cc, Cc);
  wcvt_k<<<(int)((CC  + 255) / 256), 256, 0, stream>>>(Wv, Wvt, Cc, Cc);
  wcvt_k<<<(int)((CC  + 255) / 256), 256, 0, stream>>>(Wo, Wot, Cc, Cc);
  wcvt_k<<<(int)((C4C + 255) / 256), 256, 0, stream>>>(W1, W1t, Cc, 4 * Cc);
  wcvt_k<<<(int)((C4C + 255) / 256), 256, 0, stream>>>(W2, W2t, 4 * Cc, Cc);

  const dim3 blk(128);
  const dim3 gC(Mrows / 128, Cc / 64);        // 64 x 12
  const dim3 g4C(Mrows / 128, (4 * Cc) / 64); // 64 x 48

  // 2) attention branch
  ln_k<<<Mrows, 256, 0, stream>>>(x, g1, b1, h1b);
  gemm_k<Cc, Cc, 0><<<gC, blk, 0, stream>>>(h1b, Wqt, bq, nullptr, nullptr, qbf);
  gemm_k<Cc, Cc, 0><<<gC, blk, 0, stream>>>(h1b, Wkt, bk, nullptr, nullptr, kbf);
  gemm_k<Cc, Cc, 1><<<gC, blk, 0, stream>>>(h1b, Wvt, bv, nullptr, nullptr, vtb);
  attn_k<<<dim3(Tt / 64, Bsz * Hh), blk, 0, stream>>>(qbf, kbf, vtb, atb);
  gemm_k<Cc, Cc, 2><<<gC, blk, 0, stream>>>(atb, Wot, bo, x, x2, nullptr);

  // 3) MLP branch
  ln_k<<<Mrows, 256, 0, stream>>>(x2, g2, b2, h2b);
  gemm_k<4 * Cc, Cc, 3><<<g4C, blk, 0, stream>>>(h2b, W1t, bm1, nullptr, nullptr, mb);
  gemm_k<Cc, 4 * Cc, 2><<<gC, blk, 0, stream>>>(mb, W2t, bm2, x2, out, nullptr);
}
